// MultiHeadAttention_75299366634035
// MI455X (gfx1250) — matrix operand order repaired and checked by reading, not compiled, not run on hardware
//
#include <hip/hip_runtime.h>
#include <stdint.h>

// ---------------------------------------------------------------------------
// MI455X (gfx1250) multi-head attention, bf16 WMMA path.
// ~137 GFLOP vs ~200MB traffic (9us @23.3TB/s) -> matrix-core bound, so all
// GEMM-shaped work runs on v_wmma_f32_16x16x32_bf16.  Tile staging uses the
// CDNA5 async global->LDS engine (ASYNCcnt), softmax reductions use DPP16
// butterflies, and row-sums are computed by WMMA against an all-ones fragment.
// ---------------------------------------------------------------------------

typedef __bf16    bf16;
typedef __attribute__((ext_vector_type(16))) __bf16 v16bf;
typedef __attribute__((ext_vector_type(8)))  float  v8f;
typedef __attribute__((ext_vector_type(4)))  unsigned int u32x4;

#define D_MODEL 1024
#define NUM_HEADS 16
#define D_K 64
#define BATCH 4
#define SEQ 2048

#if defined(__has_builtin)
#if __has_builtin(__builtin_amdgcn_update_dpp)
#define USE_DPP 1
#endif
#endif

__device__ __forceinline__ v8f vzero8() {
  v8f z;
#pragma unroll
  for (int i = 0; i < 8; ++i) z[i] = 0.0f;
  return z;
}

#ifdef USE_DPP
// dpp_ctrl must be a literal constant at the builtin call -> template param.
template <int CTRL>
__device__ __forceinline__ float dpp_mov(float x) {
  return __builtin_bit_cast(
      float, __builtin_amdgcn_update_dpp(0, __builtin_bit_cast(int, x), CTRL,
                                         0xF, 0xF, true));
}
// Full max-reduction across a DPP row of 16 lanes (halves stay independent).
__device__ __forceinline__ float red_max16(float x) {
  x = fmaxf(x, dpp_mov<0xB1>(x));   // quad_perm [1,0,3,2]  (xor 1)
  x = fmaxf(x, dpp_mov<0x4E>(x));   // quad_perm [2,3,0,1]  (xor 2)
  x = fmaxf(x, dpp_mov<0x141>(x));  // row_half_mirror      (xor 7)
  x = fmaxf(x, dpp_mov<0x140>(x));  // row_mirror           (xor 15)
  return x;
}
#else
__device__ __forceinline__ float red_max16(float x) {
#pragma unroll
  for (int mk = 1; mk < 16; mk <<= 1) x = fmaxf(x, __shfl_xor(x, mk, 32));
  return x;
}
#endif

// Async global->LDS copy, 16B per lane (GLOBAL_LOAD_ASYNC_TO_LDS_B128, GV
// mode).  lds_addr = low 32 bits of the generic shared pointer (ISA 10.2).
// Tracked by ASYNCcnt.
__device__ __forceinline__ void async_copy_b128(uint32_t lds_addr,
                                                uint64_t gaddr) {
  asm volatile("global_load_async_to_lds_b128 %0, %1, off"
               :: "v"(lds_addr), "v"(gaddr)
               : "memory");
}
__device__ __forceinline__ void wait_async0() {
  asm volatile("s_wait_asynccnt 0" ::: "memory");
}

// A-fragment (16x32, bf16): per lane, octet at +0 and octet at +16 elements.
__device__ __forceinline__ v16bf load_frag_a(const bf16* base) {
  union { u32x4 q[2]; v16bf v; } u;
  u.q[0] = *(const u32x4*)(base);
  u.q[1] = *(const u32x4*)(base + 16);
  return u.v;
}

// B-fragment (32x16, bf16): per lane, octet at +0 and octet at +16 elements.
__device__ __forceinline__ v16bf load_frag_b(const bf16* base) {
  union { u32x4 q[2]; v16bf v; } u;
  u.q[0] = *(const u32x4*)(base);
  u.q[1] = *(const u32x4*)(base + 16);
  return u.v;
}

__device__ __forceinline__ v8f wmma_bf16(v16bf a, v16bf b, v8f c) {
  return __builtin_amdgcn_wmma_f32_16x16x32_bf16(
      /*neg_a=*/false, a, /*neg_b=*/false, b,
      /*c_mod=*/(short)0, c, /*reuse_a=*/false, /*reuse_b=*/false);
}

// ---------------------------------------------------------------------------
// fp32 -> bf16 conversion (4 elements / thread)
// ---------------------------------------------------------------------------
__global__ __launch_bounds__(256) void convert_f32_bf16_kernel(
    const float* __restrict__ x, uint16_t* __restrict__ yp, int n) {
  bf16* y = (bf16*)yp;
  int base = (blockIdx.x * 256 + threadIdx.x) * 4;
  if (base + 3 >= n) return;
#pragma unroll
  for (int e = 0; e < 4; ++e) y[base + e] = (bf16)x[base + e];
}

// ---------------------------------------------------------------------------
// Transpose + convert weight: W fp32 [K][N] row-major -> Wt bf16 [N][K].
// ---------------------------------------------------------------------------
__global__ __launch_bounds__(256) void transpose_convert_kernel(
    const float* __restrict__ W, uint16_t* __restrict__ Wtp, int N, int K) {
  bf16* Wt = (bf16*)Wtp;
  __shared__ float tile[32][33];
  int bk = blockIdx.x * 32;  // k-origin
  int bn = blockIdx.y * 32;  // n-origin
  int tx = threadIdx.x, ty = threadIdx.y;
#pragma unroll
  for (int i = 0; i < 32; i += 8)
    tile[ty + i][tx] = W[(size_t)(bk + ty + i) * N + (bn + tx)];
  __syncthreads();
#pragma unroll
  for (int i = 0; i < 32; i += 8)
    Wt[(size_t)(bn + ty + i) * K + (bk + tx)] = (bf16)tile[tx][ty + i];
}

// ---------------------------------------------------------------------------
// bf16 GEMM: C[M,N] = alpha*(A[M,K] * Bt[N,K]^T + bias).  Block tile 128x128,
// 8 waves each 32x64 (2x4 WMMA accs), K-chunk 64, double-buffered LDS filled
// by async global->LDS copies overlapped with WMMA compute.
// ---------------------------------------------------------------------------
template <bool OUT_F32>
__global__ __launch_bounds__(256) void gemm_bf16_kernel(
    const uint16_t* __restrict__ Ap, const uint16_t* __restrict__ Btp,
    const float* __restrict__ bias, void* __restrict__ Cout,
    int M, int N, int K, float alpha) {
  const bf16* A  = (const bf16*)Ap;
  const bf16* Bt = (const bf16*)Btp;

  constexpr int GM = 128, GN = 128, GK = 64;
  __shared__ bf16 As[2][GM * GK];
  __shared__ bf16 Bs[2][GN * GK];

  const int tid  = threadIdx.x;
  const int wid  = tid >> 5;
  const int lane = tid & 31;
  const int half = lane >> 4;
  const int lr   = lane & 15;
  const int m_blk = blockIdx.y * GM;
  const int n_blk = blockIdx.x * GN;
  const int wm = (wid & 3) * 32;   // wave row offset in tile
  const int wn = (wid >> 2) * 64;  // wave col offset in tile

  const uint32_t asBase = (uint32_t)(uintptr_t)&As[0][0];
  const uint32_t bsBase = (uint32_t)(uintptr_t)&Bs[0][0];

  v8f acc[2][4];
#pragma unroll
  for (int i = 0; i < 2; ++i)
#pragma unroll
    for (int j = 0; j < 4; ++j) acc[i][j] = vzero8();

  const int KC = K / GK;
  const int row0 = tid >> 3;  // for prefetch address

  // Stage chunk 0 (async, direct to LDS).
#pragma unroll
  for (int i = 0; i < 4; ++i) {
    int j = tid + i * 256, row = j >> 3, c = j & 7;
    async_copy_b128(asBase + (uint32_t)(row * GK + c * 8) * 2,
                    (uint64_t)(uintptr_t)(A + (size_t)(m_blk + row) * K + c * 8));
    async_copy_b128(bsBase + (uint32_t)(row * GK + c * 8) * 2,
                    (uint64_t)(uintptr_t)(Bt + (size_t)(n_blk + row) * K + c * 8));
  }
  wait_async0();
  __syncthreads();

  for (int kc = 0; kc < KC; ++kc) {
    const int buf = kc & 1;
    if (kc + 1 < KC) {
      // Issue async copies for the next chunk into the other buffer; they
      // drain into LDS while this chunk's WMMAs execute.
      const uint32_t dstOff = (uint32_t)((buf ^ 1) * GM * GK) * 2;
#pragma unroll
      for (int i = 0; i < 4; ++i) {
        int j = tid + i * 256, row = j >> 3, c = j & 7;
        async_copy_b128(asBase + dstOff + (uint32_t)(row * GK + c * 8) * 2,
                        (uint64_t)(uintptr_t)(A + (size_t)(m_blk + row) * K +
                                              (size_t)(kc + 1) * GK + c * 8));
        async_copy_b128(bsBase + dstOff + (uint32_t)(row * GK + c * 8) * 2,
                        (uint64_t)(uintptr_t)(Bt + (size_t)(n_blk + row) * K +
                                              (size_t)(kc + 1) * GK + c * 8));
      }
      if (kc + 2 < KC) {  // global_prefetch_b8 two chunks ahead
        __builtin_prefetch(A  + (size_t)(m_blk + row0) * K + (kc + 2) * GK, 0, 3);
        __builtin_prefetch(Bt + (size_t)(n_blk + row0) * K + (kc + 2) * GK, 0, 3);
      }
    }
    // Compute on current buffer: 2 k-steps of 32, 8 WMMAs each.
#pragma unroll
    for (int ks = 0; ks < GK; ks += 32) {
      v16bf af[2];
      af[0] = load_frag_a(&As[buf][(wm + 0  + lr) * GK + ks + half * 8]);
      af[1] = load_frag_a(&As[buf][(wm + 16 + lr) * GK + ks + half * 8]);
      v16bf bfr[4];
#pragma unroll
      for (int nt = 0; nt < 4; ++nt)
        bfr[nt] = load_frag_b(&Bs[buf][(wn + nt * 16 + lr) * GK + ks + half * 8]);
#pragma unroll
      for (int mt = 0; mt < 2; ++mt)
#pragma unroll
        for (int nt = 0; nt < 4; ++nt)
          acc[mt][nt] = wmma_bf16(af[mt], bfr[nt], acc[mt][nt]);
    }
    if (kc + 1 < KC) wait_async0();
    __syncthreads();
  }

  // Epilogue: C layout -> row M = j + half*8, col N = lr (per 16x16 tile).
#pragma unroll
  for (int mt = 0; mt < 2; ++mt) {
#pragma unroll
    for (int nt = 0; nt < 4; ++nt) {
      const int col = n_blk + wn + nt * 16 + lr;
      const float bv = bias[col];
#pragma unroll
      for (int j = 0; j < 8; ++j) {
        const int row = m_blk + wm + mt * 16 + j + half * 8;
        const float v = (acc[mt][nt][j] + bv) * alpha;
        if (OUT_F32)
          ((float*)Cout)[(size_t)row * N + col] = v;
        else
          ((bf16*)Cout)[(size_t)row * N + col] = (bf16)v;
      }
    }
  }
}

// ---------------------------------------------------------------------------
// Flash attention: grid (S/128, B*H), block 256 (8 waves).  Each wave owns 16
// query rows; iterate 64-key blocks with K staged via async global->LDS and V
// manually transposed into LDS; online softmax (DPP max, WMMA row-sum); P
// rearranged C->A layout through per-wave LDS.  Q pre-scaled by 1/sqrt(dk).
// ---------------------------------------------------------------------------
__global__ __launch_bounds__(256) void flash_attn_kernel(
    const uint16_t* __restrict__ Qp, const uint16_t* __restrict__ Kp,
    const uint16_t* __restrict__ Vp, uint16_t* __restrict__ Ctxp) {
  const bf16* Q = (const bf16*)Qp;
  const bf16* Kb = (const bf16*)Kp;
  const bf16* Vb = (const bf16*)Vp;
  bf16* Ctx = (bf16*)Ctxp;

  __shared__ bf16 Kt[64 * 64];        // [key][d]
  __shared__ bf16 Vt[64 * 64];        // [d][key] (transposed)
  __shared__ bf16 Pt[8 * 16 * 64];    // per-wave P tiles

  const int tid = threadIdx.x, wid = tid >> 5, lane = tid & 31;
  const int half = lane >> 4, lr = lane & 15;
  const int bh = blockIdx.y;
  const int b = bh >> 4, h = bh & 15;
  const int q0 = blockIdx.x * 128 + wid * 16;  // query row within sequence
  const int hc = h * D_K;
  const size_t rowQ = (size_t)b * SEQ + q0;
  const uint32_t ktBase = (uint32_t)(uintptr_t)&Kt[0];

  // Q A-fragments (held in registers for the whole kernel).
  v16bf aq[2];
  {
    const bf16* qp = Q + (rowQ + lr) * D_MODEL + hc;
    aq[0] = load_frag_a(qp + 0  + half * 8);
    aq[1] = load_frag_a(qp + 32 + half * 8);
  }

  // All-ones B fragment: P x ones gives per-row sums in C layout.
  v16bf ones;
#pragma unroll
  for (int i = 0; i < 16; ++i) ones[i] = (bf16)1.0f;

  float m_run[8], l_run[8];
  v8f oacc[4];
#pragma unroll
  for (int j = 0; j < 8; ++j) { m_run[j] = -3.0e38f; l_run[j] = 0.0f; }
#pragma unroll
  for (int nt = 0; nt < 4; ++nt) oacc[nt] = vzero8();

  bf16* Pw = &Pt[wid * 16 * 64];

  for (int kb = 0; kb < SEQ / 64; ++kb) {
    const size_t krow = (size_t)b * SEQ + kb * 64;
    // Stage K via async global->LDS; V via manual transposed stores.
#pragma unroll
    for (int it = 0; it < 2; ++it) {
      int j = tid + it * 256;
      int key = j >> 3, c = j & 7;
      async_copy_b128(ktBase + (uint32_t)(key * 64 + c * 8) * 2,
                      (uint64_t)(uintptr_t)(Kb + (krow + key) * D_MODEL + hc + c * 8));
      u32x4 vv = *(const u32x4*)(Vb + (krow + key) * D_MODEL + hc + c * 8);
      const bf16* vp = (const bf16*)&vv;
#pragma unroll
      for (int e = 0; e < 8; ++e) Vt[(c * 8 + e) * 64 + key] = vp[e];
    }
    wait_async0();
    __syncthreads();

    // scores = (Q/sqrt(dk)) * K^T  (B-fragment: n = key, k = d)
    v8f sacc[4];
#pragma unroll
    for (int nt = 0; nt < 4; ++nt) sacc[nt] = vzero8();
#pragma unroll
    for (int ks = 0; ks < 2; ++ks) {
#pragma unroll
      for (int nt = 0; nt < 4; ++nt) {
        v16bf bk = load_frag_b(&Kt[(nt * 16 + lr) * 64 + ks * 32 + half * 8]);
        sacc[nt] = wmma_bf16(aq[ks], bk, sacc[nt]);
      }
    }

    // Online softmax: DPP-based row max (VALU only), exp, O rescale.
    float corr[8];
#pragma unroll
    for (int j = 0; j < 8; ++j) {
      float rm = fmaxf(fmaxf(sacc[0][j], sacc[1][j]),
                       fmaxf(sacc[2][j], sacc[3][j]));
      rm = red_max16(rm);
      const float mn = fmaxf(m_run[j], rm);
      corr[j] = __expf(m_run[j] - mn);
      m_run[j] = mn;
#pragma unroll
      for (int nt = 0; nt < 4; ++nt) sacc[nt][j] = __expf(sacc[nt][j] - mn);
#pragma unroll
      for (int nt = 0; nt < 4; ++nt) oacc[nt][j] *= corr[j];
    }

    // P: C-layout -> per-wave LDS -> A-layout fragments (wave-local).
#pragma unroll
    for (int nt = 0; nt < 4; ++nt)
#pragma unroll
      for (int j = 0; j < 8; ++j)
        Pw[(j + half * 8) * 64 + nt * 16 + lr] = (bf16)sacc[nt][j];
    asm volatile("s_wait_dscnt 0" ::: "memory");

    v16bf ap[2];
    ap[0] = load_frag_a(&Pw[lr * 64 + 0  + half * 8]);
    ap[1] = load_frag_a(&Pw[lr * 64 + 32 + half * 8]);

    // Row sums via WMMA against the ones fragment (replaces shuffle reduce).
    v8f ssum = vzero8();
    ssum = wmma_bf16(ap[0], ones, ssum);
    ssum = wmma_bf16(ap[1], ones, ssum);

    // O += P * V  (B-fragment from Vt: n = d, k = key)
#pragma unroll
    for (int ks = 0; ks < 2; ++ks) {
#pragma unroll
      for (int nt = 0; nt < 4; ++nt) {
        v16bf bv = load_frag_b(&Vt[(nt * 16 + lr) * 64 + ks * 32 + half * 8]);
        oacc[nt] = wmma_bf16(ap[ks], bv, oacc[nt]);
      }
    }

#pragma unroll
    for (int j = 0; j < 8; ++j) l_run[j] = l_run[j] * corr[j] + ssum[j];

    __syncthreads();
  }

  // Normalize and write context (bf16).
#pragma unroll
  for (int j = 0; j < 8; ++j) {
    const float inv = 1.0f / l_run[j];
    const int row = q0 + j + half * 8;
#pragma unroll
    for (int nt = 0; nt < 4; ++nt)
      Ctx[((size_t)b * SEQ + row) * D_MODEL + hc + nt * 16 + lr] =
          (bf16)(oacc[nt][j] * inv);
  }
}

// ---------------------------------------------------------------------------
// Host-side orchestration.
// ---------------------------------------------------------------------------
extern "C" void kernel_launch(void* const* d_in, const int* in_sizes, int n_in,
                              void* d_out, int out_size, void* d_ws, size_t ws_size,
                              hipStream_t stream) {
  const float* q_input = (const float*)d_in[0];
  // NOTE: reference quirk — K and V projections also consume q_input.
  const float* Wq = (const float*)d_in[3];
  const float* bq = (const float*)d_in[4];
  const float* Wk = (const float*)d_in[5];
  const float* bk = (const float*)d_in[6];
  const float* Wv = (const float*)d_in[7];
  const float* bv = (const float*)d_in[8];
  const float* Wo = (const float*)d_in[9];
  const float* bo = (const float*)d_in[10];
  float* out = (float*)d_out;

  const int M = BATCH * SEQ;        // 8192
  const int D = D_MODEL;            // 1024
  const size_t xBytes = (size_t)M * D * sizeof(uint16_t);
  const size_t wBytes = (size_t)D * D * sizeof(uint16_t);

  char* ws = (char*)d_ws;
  size_t off = 0;
  auto carve = [&](size_t bytes) {
    void* p = ws + off;
    off += (bytes + 255) & ~(size_t)255;
    return p;
  };
  uint16_t* Xb  = (uint16_t*)carve(xBytes);
  uint16_t* Wqt = (uint16_t*)carve(wBytes);
  uint16_t* Wkt = (uint16_t*)carve(wBytes);
  uint16_t* Wvt = (uint16_t*)carve(wBytes);
  uint16_t* Wot = (uint16_t*)carve(wBytes);
  uint16_t* Qb  = (uint16_t*)carve(xBytes);
  uint16_t* Kb  = (uint16_t*)carve(xBytes);
  uint16_t* Vb  = (uint16_t*)carve(xBytes);
  uint16_t* Ctx = (uint16_t*)carve(xBytes);

  const float kScale = 0.125f;  // 1/sqrt(D_K)

  // 1) Input conversion fp32 -> bf16.
  {
    int n = M * D;
    convert_f32_bf16_kernel<<<n / (256 * 4), 256, 0, stream>>>(q_input, Xb, n);
  }
  // 2) Weight transpose + convert: W[K][N] -> Wt[N][K] bf16.
  {
    dim3 blk(32, 8), grd(D / 32, D / 32);
    transpose_convert_kernel<<<grd, blk, 0, stream>>>(Wq, Wqt, D, D);
    transpose_convert_kernel<<<grd, blk, 0, stream>>>(Wk, Wkt, D, D);
    transpose_convert_kernel<<<grd, blk, 0, stream>>>(Wv, Wvt, D, D);
    transpose_convert_kernel<<<grd, blk, 0, stream>>>(Wo, Wot, D, D);
  }
  // 3) Q/K/V projections (bf16 out; Q pre-scaled by 1/sqrt(dk)).
  {
    dim3 grd(D / 128, M / 128);
    gemm_bf16_kernel<false><<<grd, 256, 0, stream>>>(Xb, Wqt, bq, Qb, M, D, D, kScale);
    gemm_bf16_kernel<false><<<grd, 256, 0, stream>>>(Xb, Wkt, bk, Kb, M, D, D, 1.0f);
    gemm_bf16_kernel<false><<<grd, 256, 0, stream>>>(Xb, Wvt, bv, Vb, M, D, D, 1.0f);
  }
  // 4) Flash attention per (b,h).
  {
    dim3 grd(SEQ / 128, BATCH * NUM_HEADS);
    flash_attn_kernel<<<grd, 256, 0, stream>>>(Qb, Kb, Vb, Ctx);
  }
  // 5) Output projection (fp32 out + bias).
  {
    dim3 grd(D / 128, M / 128);
    gemm_bf16_kernel<true><<<grd, 256, 0, stream>>>(Ctx, Wot, bo, out, M, D, D, 1.0f);
  }
  (void)in_sizes; (void)n_in; (void)out_size; (void)ws_size;
}
